// BiLSTMCRFSequenceTaggingModel_63780264345883
// MI455X (gfx1250) — matrix operand order repaired and checked by reading, not compiled, not run on hardware
//
#include <hip/hip_runtime.h>
#include <hip/hip_bf16.h>

// ---------------------------------------------------------------------------
// BiLSTM-CRF for MI455X (gfx1250, wave32, WMMA).
//   V=50000 E=300 H=256 C=20 B=128 T=512
// Pipeline:
//   1. embed gather -> f16, K padded 300->320
//   2. f32->f16 weight conversion (+ bias sum)
//   3. WMMA GEMM: xg[dir][T][B][4H] = X @ w_ih^T + b   (v_wmma_f32_16x16x32_f16)
//   4. recurrent scan: per-step WMMA h @ w_hh^T, h in LDS, c in acc registers
//   5. FC emissions (C=20, VALU)
//   6. CRF forward algorithm + gold score, atomicAdd into scalar loss
// ---------------------------------------------------------------------------

typedef __attribute__((ext_vector_type(16))) _Float16 v16h;
typedef __attribute__((ext_vector_type(8)))  float    v8f;

constexpr int Bn  = 128;
constexpr int Tn  = 512;
constexpr int En  = 300;
constexpr int Hn  = 256;
constexpr int Cn  = 20;
constexpr int BT  = Bn * Tn;      // 65536
constexpr int G4H = 4 * Hn;       // 1024
constexpr int KP0 = 320;          // E padded to mult of 32
constexpr int KP1 = 2 * Hn;       // 512

union FragU {
    v16h v;
    uint4 q[2];
};

// A fragment: 16x32 f16, rows = m-tile, K-contiguous row-major source.
// lanes 0-15: halves 0-7 -> K k0+0..7, halves 8-15 -> K k0+16..23
// lanes 16-31: halves 0-7 -> K k0+8..15, halves 8-15 -> K k0+24..31
__device__ __forceinline__ v16h load_frag_a(const _Float16* __restrict__ base,
                                            int ld, int row0, int k0, int lane) {
    int m = lane & 15;
    const _Float16* p = base + (size_t)(row0 + m) * ld + k0 + ((lane & 16) ? 8 : 0);
    FragU f;
    f.q[0] = *(const uint4*)(p);        // K +0..7
    f.q[1] = *(const uint4*)(p + 16);   // K +16..23
    return f.v;
}

// B fragment: 32x16 f16 where B[k][n] = W[n0+n][k], W row-major [N, ld].
// lanes 0-15: col n, halves 0-7 -> K k0+0..7, 8-15 -> K k0+16..23 ; lanes 16-31: K k0+8..15, k0+24..31
__device__ __forceinline__ v16h load_frag_b(const _Float16* __restrict__ W,
                                            int ld, int n0, int k0, int lane) {
    int n = lane & 15;
    const _Float16* p = W + (size_t)(n0 + n) * ld + k0 + ((lane & 16) ? 8 : 0);
    FragU f;
    f.q[0] = *(const uint4*)(p);
    f.q[1] = *(const uint4*)(p + 16);
    return f.v;
}

__device__ __forceinline__ float sigf(float x) { return 1.0f / (1.0f + __expf(-x)); }

// ---------------------------------------------------------------------------
// prep kernels
// ---------------------------------------------------------------------------
__global__ void zero_out_k(float* out) {
    if (blockIdx.x == 0 && threadIdx.x == 0) out[0] = 0.0f;
}

__global__ void embed_gather_k(const int* __restrict__ ids,
                               const float* __restrict__ emb,
                               _Float16* __restrict__ x) {
    size_t i = (size_t)blockIdx.x * blockDim.x + threadIdx.x;
    if (i >= (size_t)BT * KP0) return;
    int bt = (int)(i / KP0);
    int e  = (int)(i % KP0);
    float v = (e < En) ? emb[(size_t)ids[bt] * En + e] : 0.0f;
    x[i] = (_Float16)v;
}

__global__ void cvt_pad_f16_k(const float* __restrict__ src,
                              _Float16* __restrict__ dst,
                              int rows, int kin, int kout) {
    size_t i = (size_t)blockIdx.x * blockDim.x + threadIdx.x;
    if (i >= (size_t)rows * kout) return;
    int r = (int)(i / kout);
    int k = (int)(i % kout);
    dst[i] = (_Float16)((k < kin) ? src[(size_t)r * kin + k] : 0.0f);
}

__global__ void bias_sum_k(const float* __restrict__ a,
                           const float* __restrict__ b,
                           float* __restrict__ o, int n) {
    int i = blockIdx.x * blockDim.x + threadIdx.x;
    if (i < n) o[i] = a[i] + b[i];
}

// ---------------------------------------------------------------------------
// Input-projection WMMA GEMM:
//   Y[dir][T][B][4H] (f32) = X[b*T+t][Kp] (f16) @ W[dir][4H][Kp]^T + bias[dir]
// block = 256 threads (8 waves), tile 128(M) x 64(N), wave tile 32x32.
// ---------------------------------------------------------------------------
__global__ void wmma_gemm_xg_k(const _Float16* __restrict__ X,
                               const _Float16* __restrict__ W,
                               const float* __restrict__ bias,
                               float* __restrict__ Y,
                               int Kp) {
    constexpr int N = G4H;
    int lane = threadIdx.x & 31;
    int wave = threadIdx.x >> 5;
    int dir  = blockIdx.z;
    int m0 = blockIdx.x * 128 + (wave & 3) * 32;
    int n0 = blockIdx.y * 64 + (wave >> 2) * 32;
    const _Float16* Wd = W + (size_t)dir * N * Kp;

    v8f acc00 = {}, acc01 = {}, acc10 = {}, acc11 = {};
    for (int k0 = 0; k0 < Kp; k0 += 32) {
        v16h a0 = load_frag_a(X, Kp, m0, k0, lane);
        v16h a1 = load_frag_a(X, Kp, m0 + 16, k0, lane);
        v16h b0 = load_frag_b(Wd, Kp, n0, k0, lane);
        v16h b1 = load_frag_b(Wd, Kp, n0 + 16, k0, lane);
        acc00 = __builtin_amdgcn_wmma_f32_16x16x32_f16(false, a0, false, b0, (short)0, acc00, false, false);
        acc01 = __builtin_amdgcn_wmma_f32_16x16x32_f16(false, a0, false, b1, (short)0, acc01, false, false);
        acc10 = __builtin_amdgcn_wmma_f32_16x16x32_f16(false, a1, false, b0, (short)0, acc10, false, false);
        acc11 = __builtin_amdgcn_wmma_f32_16x16x32_f16(false, a1, false, b1, (short)0, acc11, false, false);
    }

    float* Yd = Y + (size_t)dir * Tn * Bn * N;
    int hi = (lane & 16) ? 8 : 0;
    int nc = lane & 15;
#pragma unroll
    for (int mi = 0; mi < 2; ++mi) {
#pragma unroll
        for (int ni = 0; ni < 2; ++ni) {
            v8f acc = (mi == 0) ? ((ni == 0) ? acc00 : acc01)
                                : ((ni == 0) ? acc10 : acc11);
            int n = n0 + ni * 16 + nc;
            float bv = bias[(size_t)dir * N + n];
#pragma unroll
            for (int r = 0; r < 8; ++r) {
                int m = m0 + mi * 16 + hi + r;     // row in [B*T], b-major
                int b = m >> 9;                    // /T
                int t = m & (Tn - 1);              // %T
                Yd[((size_t)t * Bn + b) * N + n] = acc[r] + bv;
            }
        }
    }
}

// ---------------------------------------------------------------------------
// Recurrent scan (one LSTM direction chunk):
// grid.x = B/16 batch chunks, grid.y = dir. 256 threads = 8 waves.
// wave w owns hidden cols [w*32, w*32+32): 4 gates x 2 col-tiles = 8 accs.
// h lives in LDS (16 x 256 f16); c lives in accumulator registers.
// Per step: 8 K-iters x 8 tiles = 64 v_wmma per wave.
// ---------------------------------------------------------------------------
__global__ void lstm_scan_k(const float* __restrict__ xg,      // [2][T][B][4H]
                            const _Float16* __restrict__ Whh,  // [2][4H][H] f16
                            _Float16* __restrict__ hcat) {     // [B][T][2H]
    constexpr int N = G4H;
    constexpr int K = Hn;
    __shared__ _Float16 hbuf[16 * Hn];                          // 8 KB

    int lane = threadIdx.x & 31;
    int wave = threadIdx.x >> 5;
    int d  = blockIdx.y;
    int b0 = blockIdx.x * 16;
    int j0 = wave * 32;
    const _Float16* Wd  = Whh + (size_t)d * N * K;
    const float*    xgd = xg + (size_t)d * Tn * Bn * N;

    for (int i = threadIdx.x; i < 16 * Hn; i += blockDim.x) hbuf[i] = (_Float16)0.0f;

    int hi = (lane & 16) ? 8 : 0;
    int nc = lane & 15;
    v8f c0 = {}, c1 = {};
    __syncthreads();

    for (int s = 0; s < Tn; ++s) {
        int t = d ? (Tn - 1 - s) : s;

        // init accumulators with precomputed input-projection gates
        v8f acc[4][2];
#pragma unroll
        for (int g = 0; g < 4; ++g) {
#pragma unroll
            for (int ni = 0; ni < 2; ++ni) {
                int n = g * Hn + j0 + ni * 16 + nc;
#pragma unroll
                for (int r = 0; r < 8; ++r) {
                    int m = hi + r;
                    acc[g][ni][r] = xgd[((size_t)t * Bn + b0 + m) * N + n];
                }
            }
        }

        // CDNA5 prefetch of next timestep's gate tile (global_prefetch_b8)
        if (s + 1 < Tn) {
            int tn = d ? (Tn - 2 - s) : (s + 1);
            __builtin_prefetch(xgd + ((size_t)tn * Bn + b0 + hi) * N + j0, 0, 1);
        }

        // recurrent GEMM: gates += h @ Whh^T
        for (int kk = 0; kk < K; kk += 32) {
            int m = lane & 15;
            const _Float16* p = hbuf + m * Hn + kk + ((lane & 16) ? 8 : 0);
            FragU fa;
            fa.q[0] = *(const uint4*)(p);
            fa.q[1] = *(const uint4*)(p + 16);
            v16h a = fa.v;
#pragma unroll
            for (int g = 0; g < 4; ++g) {
#pragma unroll
                for (int ni = 0; ni < 2; ++ni) {
                    v16h b = load_frag_b(Wd, K, g * Hn + j0 + ni * 16, kk, lane);
                    acc[g][ni] = __builtin_amdgcn_wmma_f32_16x16x32_f16(
                        false, a, false, b, (short)0, acc[g][ni], false, false);
                }
            }
        }
        __syncthreads();   // all reads of hbuf (old h) complete

        // gate nonlinearities, state update, write new h
#pragma unroll
        for (int ni = 0; ni < 2; ++ni) {
#pragma unroll
            for (int r = 0; r < 8; ++r) {
                float iv = sigf(acc[0][ni][r]);
                float fv = sigf(acc[1][ni][r]);
                float gv = tanhf(acc[2][ni][r]);
                float ov = sigf(acc[3][ni][r]);
                float cp = ni ? c1[r] : c0[r];
                float cnv = fv * cp + iv * gv;
                if (ni) c1[r] = cnv; else c0[r] = cnv;
                float hv = ov * tanhf(cnv);
                int m = hi + r;
                int j = j0 + ni * 16 + nc;
                hbuf[m * Hn + j] = (_Float16)hv;
                hcat[(((size_t)(b0 + m)) * Tn + t) * (2 * Hn) + d * Hn + j] = (_Float16)hv;
            }
        }
        __syncthreads();   // new h visible before next step's reads
    }
}

// ---------------------------------------------------------------------------
// FC: emissions[b][t][c] = hcat[b][t][:] . fc_w[c][:] + fc_b[c]   (C=20)
// ---------------------------------------------------------------------------
__global__ void fc_emissions_k(const _Float16* __restrict__ hcat,
                               const float* __restrict__ fcw,
                               const float* __restrict__ fcb,
                               float* __restrict__ em) {
    int idx = blockIdx.x * blockDim.x + threadIdx.x;
    if (idx >= BT) return;
    const _Float16* xr = hcat + (size_t)idx * KP1;
#pragma unroll 4
    for (int c = 0; c < Cn; ++c) {
        const float* w = fcw + (size_t)c * KP1;
        float s = fcb[c];
        for (int k = 0; k < KP1; ++k) s += (float)xr[k] * w[k];
        em[(size_t)idx * Cn + c] = s;
    }
}

// ---------------------------------------------------------------------------
// CRF NLL: one wave per batch row; forward algorithm in LDS; sum via atomic.
// ---------------------------------------------------------------------------
__global__ void crf_nll_k(const float* __restrict__ em,       // [B][T][C]
                          const int* __restrict__ labels,     // [B][T]
                          const float* __restrict__ trans,    // [C][C]
                          const float* __restrict__ startT,
                          const float* __restrict__ endT,
                          float* __restrict__ out) {
    __shared__ float buf0[Cn], buf1[Cn], tr[Cn * Cn];
    int b = blockIdx.x;
    int lane = threadIdx.x;
    const float* eb = em + (size_t)b * Tn * Cn;

    for (int i = lane; i < Cn * Cn; i += 32) tr[i] = trans[i];
    if (lane < Cn) buf0[lane] = startT[lane] + eb[lane];
    __syncthreads();

    float* cur = buf0;
    float* nxt = buf1;
    for (int t = 1; t < Tn; ++t) {
        if (lane < Cn) {
            float mx = -3.4e38f;
            for (int p = 0; p < Cn; ++p) mx = fmaxf(mx, cur[p] + tr[p * Cn + lane]);
            float s = 0.0f;
            for (int p = 0; p < Cn; ++p) s += __expf(cur[p] + tr[p * Cn + lane] - mx);
            nxt[lane] = mx + __logf(s) + eb[(size_t)t * Cn + lane];
        }
        __syncthreads();
        float* tmp = cur; cur = nxt; nxt = tmp;
    }

    if (lane == 0) {
        float mx = -3.4e38f;
        for (int c = 0; c < Cn; ++c) mx = fmaxf(mx, cur[c] + endT[c]);
        float s = 0.0f;
        for (int c = 0; c < Cn; ++c) s += __expf(cur[c] + endT[c] - mx);
        float logZ = mx + __logf(s);

        const int* tg = labels + (size_t)b * Tn;
        float num = startT[tg[0]] + endT[tg[Tn - 1]];
        for (int t = 0; t < Tn; ++t) num += eb[(size_t)t * Cn + tg[t]];
        for (int t = 1; t < Tn; ++t) num += tr[tg[t - 1] * Cn + tg[t]];
        atomicAdd(out, logZ - num);   // loss = sum(logZ - gold)
    }
}

// ---------------------------------------------------------------------------
// host launcher
// ---------------------------------------------------------------------------
extern "C" void kernel_launch(void* const* d_in, const int* in_sizes, int n_in,
                              void* d_out, int out_size, void* d_ws, size_t ws_size,
                              hipStream_t stream) {
    (void)in_sizes; (void)n_in; (void)out_size; (void)ws_size;

    const int*   ids    = (const int*)d_in[0];
    const int*   labels = (const int*)d_in[1];
    const float* emb    = (const float*)d_in[2];
    const float* wih0   = (const float*)d_in[3];
    const float* whh0   = (const float*)d_in[4];
    const float* bih0   = (const float*)d_in[5];
    const float* bhh0   = (const float*)d_in[6];
    const float* wih1   = (const float*)d_in[7];
    const float* whh1   = (const float*)d_in[8];
    const float* bih1   = (const float*)d_in[9];
    const float* bhh1   = (const float*)d_in[10];
    const float* fcw    = (const float*)d_in[11];
    const float* fcb    = (const float*)d_in[12];
    const float* trans  = (const float*)d_in[13];
    const float* startT = (const float*)d_in[14];
    const float* endT   = (const float*)d_in[15];
    float* out = (float*)d_out;

    char* ws = (char*)d_ws;
    size_t off = 0;
    auto take = [&](size_t bytes) -> char* {
        char* p = ws + off;
        off = (off + bytes + 255) & ~(size_t)255;
        return p;
    };
    _Float16* x0    = (_Float16*)take((size_t)BT * KP0 * 2);          // 40 MB
    _Float16* wih0h = (_Float16*)take((size_t)2 * G4H * KP0 * 2);
    _Float16* whh0h = (_Float16*)take((size_t)2 * G4H * Hn * 2);
    _Float16* wih1h = (_Float16*)take((size_t)2 * G4H * KP1 * 2);
    _Float16* whh1h = (_Float16*)take((size_t)2 * G4H * Hn * 2);
    float*    bias0 = (float*)take((size_t)2 * G4H * 4);
    float*    bias1 = (float*)take((size_t)2 * G4H * 4);
    float*    xg    = (float*)take((size_t)2 * Tn * Bn * G4H * 4);    // 512 MB
    _Float16* hc0   = (_Float16*)take((size_t)BT * KP1 * 2);          // 64 MB
    _Float16* hc1   = (_Float16*)take((size_t)BT * KP1 * 2);          // 64 MB
    float*    emis  = (float*)take((size_t)BT * Cn * 4);

    zero_out_k<<<1, 1, 0, stream>>>(out);

    {
        size_t n = (size_t)BT * KP0;
        embed_gather_k<<<(unsigned)((n + 255) / 256), 256, 0, stream>>>(ids, emb, x0);
    }
    cvt_pad_f16_k<<<(2 * G4H * KP0 + 255) / 256, 256, 0, stream>>>(wih0, wih0h, 2 * G4H, En, KP0);
    cvt_pad_f16_k<<<(2 * G4H * Hn + 255) / 256, 256, 0, stream>>>(whh0, whh0h, 2 * G4H, Hn, Hn);
    cvt_pad_f16_k<<<(2 * G4H * KP1 + 255) / 256, 256, 0, stream>>>(wih1, wih1h, 2 * G4H, KP1, KP1);
    cvt_pad_f16_k<<<(2 * G4H * Hn + 255) / 256, 256, 0, stream>>>(whh1, whh1h, 2 * G4H, Hn, Hn);
    bias_sum_k<<<(2 * G4H + 255) / 256, 256, 0, stream>>>(bih0, bhh0, bias0, 2 * G4H);
    bias_sum_k<<<(2 * G4H + 255) / 256, 256, 0, stream>>>(bih1, bhh1, bias1, 2 * G4H);

    dim3 gemm_grid(BT / 128, G4H / 64, 2);
    dim3 scan_grid(Bn / 16, 2);

    // layer 0
    wmma_gemm_xg_k<<<gemm_grid, 256, 0, stream>>>(x0, wih0h, bias0, xg, KP0);
    lstm_scan_k<<<scan_grid, 256, 0, stream>>>(xg, whh0h, hc0);
    // layer 1
    wmma_gemm_xg_k<<<gemm_grid, 256, 0, stream>>>(hc0, wih1h, bias1, xg, KP1);
    lstm_scan_k<<<scan_grid, 256, 0, stream>>>(xg, whh1h, hc1);

    fc_emissions_k<<<(BT + 255) / 256, 256, 0, stream>>>(hc1, fcw, fcb, emis);
    crf_nll_k<<<Bn, 32, 0, stream>>>(emis, labels, trans, startT, endT, out);
}
